// Model_79310866088198
// MI455X (gfx1250) — matrix operand order repaired and checked by reading, not compiled, not run on hardware
//
#include <hip/hip_runtime.h>
#include <hip/hip_bf16.h>
#include <math.h>

// Problem sizes (match reference)
#define B_SZ 32
#define L_SZ 256
#define D_M  512
#define D_F  2048
#define N_E  8

// Tiling
#define LT   64    // L rows per block
#define FCH  32    // F chunk == WMMA K for bf16
#define NTHR 256   // 8 waves of 32

#define CHUNKS       (D_F / FCH)          // 64 f-chunks
#define FRAGB_U2     160                  // uint2 per 40B-stride fragment
#define WCHUNK_U2    (32 * FRAGB_U2)      // 5120 uint2 = 40KB per chunk image
#define WCHUNK_BYTES (WCHUNK_U2 * 8)      // 40960
#define WCHUNK_H     (WCHUNK_BYTES / 2)   // halves per chunk image
#define WS_NEED      ((size_t)2 * N_E * CHUNKS * WCHUNK_BYTES)  // ~42MB

typedef __attribute__((ext_vector_type(16))) __bf16 v16bf;
typedef __attribute__((ext_vector_type(2)))  __bf16 v2bf;
typedef __attribute__((ext_vector_type(8)))  float  v8f;

union Frag32 { v16bf v; uint4 u4[2]; };
union Frag40 { v16bf v; uint2 u2[4]; };

__device__ __forceinline__ float gelu_exact(float x) {
  return 0.5f * x * (1.0f + erff(x * 0.70710678118654752440f));
}

// A-layout half index / lane-half for K in [0,32)
__device__ __forceinline__ int a_half(int K) { return K & 15; }
__device__ __forceinline__ int a_lhalf(int K) { return (K & 16) ? 16 : 0; }

__device__ __forceinline__ v16bf load_frag32(const uint4* fragBase, int lane) {
  Frag32 f;
  const uint4* p = fragBase + lane * 2;   // 2 x ds_load_b128
  f.u4[0] = p[0];
  f.u4[1] = p[1];
  return f.v;
}

__device__ __forceinline__ v16bf load_frag40(const uint2* fragBase, int lane) {
  Frag40 f;
  const uint2* p = fragBase + lane * 5;   // 2 x ds_load_2addr_b64, conflict-free
  f.u2[0] = p[0];
  f.u2[1] = p[1];
  f.u2[2] = p[2];
  f.u2[3] = p[3];
  return f.v;
}

// ---- one-time pre-pass: convert+swizzle W1/W2 chunks to fragment-order bf16
// ws layout: [e*CHUNKS+ch] W1 images, then [N_E*CHUNKS + e*CHUNKS+ch] W2 images
__global__ __launch_bounds__(NTHR) void convert_weights_kernel(
    const float* __restrict__ W1, const float* __restrict__ W2,
    __bf16* __restrict__ wsw) {
  const int blk = blockIdx.x;               // e * CHUNKS + ch
  const int e = blk / CHUNKS, ch = blk % CHUNKS;
  const int fc = ch * FCH;
  const int tid = threadIdx.x;
  const float* W1e = W1 + (size_t)e * D_M * D_F;
  const float* W2e = W2 + (size_t)e * D_F * D_M;
  __bf16* d1 = wsw + (size_t)blk * WCHUNK_H;
  __bf16* d2 = wsw + ((size_t)N_E * CHUNKS + blk) * WCHUNK_H;

  // W1[:, fc:fc+32]: pair along K rows (dr, dr+1) -> contiguous halves h,h+1
  for (int i = tid; i < D_M * FCH / 2; i += NTHR) {
    const int drp = i >> 5, c = i & 31;
    const int dr = drp * 2;
    const float lo = W1e[(size_t)dr * D_F + fc + c];
    const float hi = W1e[(size_t)(dr + 1) * D_F + fc + c];
    const int ks = dr >> 5, kk = dr & 31;   // kk even
    const int ln = (c & 15) + ((kk & 16) ? 16 : 0);
    const int frag = ks * 2 + (c >> 4);
    v2bf pv; pv[0] = (__bf16)lo; pv[1] = (__bf16)hi;
    *(v2bf*)(d1 + frag * 640 + ln * 20 + (kk & 15)) = pv;
  }
  // W2[fc:fc+32, :]: pair along K rows (r, r+1) -> contiguous halves h,h+1
  for (int i = tid; i < FCH * D_M / 2; i += NTHR) {
    const int rp = i >> 9, c = i & 511;
    const int r = rp * 2;
    const float lo = W2e[(size_t)(fc + r) * D_M + c];
    const float hi = W2e[(size_t)(fc + r + 1) * D_M + c];
    const int ln = (c & 15) + ((r & 16) ? 16 : 0);
    v2bf pv; pv[0] = (__bf16)lo; pv[1] = (__bf16)hi;
    *(v2bf*)(d2 + (c >> 4) * 640 + ln * 20 + (r & 15)) = pv;
  }
}

template <bool USE_WS>
__global__ __launch_bounds__(NTHR) void moe_fused_kernel(
    const float* __restrict__ x, const float* __restrict__ logits,
    const int* __restrict__ masks, const float* __restrict__ W1,
    const float* __restrict__ b1, const float* __restrict__ W2,
    const float* __restrict__ b2, const __bf16* __restrict__ wsw,
    float* __restrict__ out) {
  // x A-fragments: 4 row-tiles x 16 k-steps = 64 frags, 64 uint4 each -> 64KB
  __shared__ uint4 xf[64 * 64];
  __shared__ uint4 w1f4[WCHUNK_BYTES / 16];  // 40KB chunk image
  __shared__ uint4 w2f4[WCHUNK_BYTES / 16];  // 40KB chunk image
  __shared__ uint2 hf[4 * FRAGB_U2];         // H A-fragments, 5KB
  __shared__ float gates_s[N_E];
  __shared__ float bias_s[D_M];
  uint2* w1f = (uint2*)w1f4;
  uint2* w2f = (uint2*)w2f4;

  const int tid  = threadIdx.x;
  const int lane = tid & 31;
  const int wid  = tid >> 5;
  const int b    = blockIdx.y;
  const int l0   = blockIdx.x * LT;

  // ---- gates: softmax -> mask -> top-2 -> renormalize (thread 0) ----
  if (tid == 0) {
    float p[N_E];
    float mx = -1e30f;
    for (int e = 0; e < N_E; ++e) {
      p[e] = logits[b * N_E + e];
      if (p[e] > mx) mx = p[e];
    }
    float s = 0.0f;
    for (int e = 0; e < N_E; ++e) { p[e] = expf(p[e] - mx); s += p[e]; }
    for (int e = 0; e < N_E; ++e)
      p[e] = (masks[b * N_E + e] == 1) ? (p[e] / s) : 0.0f;
    int i1 = 0;
    for (int e = 1; e < N_E; ++e) if (p[e] > p[i1]) i1 = e;
    int i2 = -1;
    for (int e = 0; e < N_E; ++e) {
      if (e == i1) continue;
      if (i2 < 0 || p[e] > p[i2]) i2 = e;
    }
    const float den = p[i1] + p[i2] + 1e-9f;
    for (int e = 0; e < N_E; ++e)
      gates_s[e] = (e == i1 || e == i2) ? (p[e] / den) : 0.0f;
  }
  __syncthreads();

  // ---- bias term: sum_e gates[e] * b2[e][d] ----
  for (int d = tid; d < D_M; d += NTHR) {
    float s = 0.0f;
    for (int e = 0; e < N_E; ++e) s += gates_s[e] * b2[e * D_M + d];
    bias_s[d] = s;
  }

  // ---- stage x tile directly into A-fragment order (float2 loads) ----
  for (int i = tid; i < LT * D_M / 2; i += NTHR) {
    const int idx = i * 2;
    const int r = idx >> 9, d = idx & 511;  // d even
    const float2 xv =
        *(const float2*)&x[((size_t)(b * L_SZ + l0 + r)) * D_M + d];
    const int rt = r >> 4, rl = r & 15;
    const int ks = d >> 5, K = d & 31;
    __bf16* dst = (__bf16*)xf + ((rt * 16 + ks) * 32 + rl + a_lhalf(K)) * 16 +
                  a_half(K);
    v2bf pv;
    pv[0] = (__bf16)xv.x;
    pv[1] = (__bf16)xv.y;
    *(v2bf*)dst = pv;
  }
  __syncthreads();

  const v8f vzero = {};
  v8f acc[16];
#pragma unroll
  for (int t = 0; t < 16; ++t) acc[t] = vzero;

  const int m1   = wid >> 1;  // GEMM1: this wave's H row tile (0..3)
  const int c1   = wid & 1;   // GEMM1: this wave's H col tile (0..1)
  const int mrow = wid & 3;   // GEMM2: output row tile (0..3)
  const int cgrp = wid >> 2;  // GEMM2: output col half (0..1)

  for (int e = 0; e < N_E; ++e) {
    const float g = gates_s[e];
    if (g == 0.0f) continue;  // uniform per block -> EXEC stays all-ones
    const float* W1e = W1 + (size_t)e * D_M * D_F;
    const float* W2e = W2 + (size_t)e * D_F * D_M;
    const float* b1e = b1 + e * D_F;

    for (int ch = 0; ch < CHUNKS; ++ch) {
      const int fc = ch * FCH;
      if (USE_WS) {
        // ---- fast path: straight b128 copy of pre-swizzled bf16 images ----
        const uint4* s1 = (const uint4*)(wsw + (size_t)(e * CHUNKS + ch) * WCHUNK_H);
        const uint4* s2 = (const uint4*)(wsw + ((size_t)N_E * CHUNKS + e * CHUNKS + ch) * WCHUNK_H);
        if (ch + 1 < CHUNKS) {  // gfx1250 global_prefetch_b8 hint
          __builtin_prefetch(s1 + WCHUNK_BYTES / 16 + tid * 10, 0, 3);
          __builtin_prefetch(s2 + WCHUNK_BYTES / 16 + tid * 10, 0, 3);
        }
#pragma unroll
        for (int i = 0; i < WCHUNK_BYTES / 16 / NTHR; ++i) {  // 10 per thread
          w1f4[i * NTHR + tid] = s1[i * NTHR + tid];
          w2f4[i * NTHR + tid] = s2[i * NTHR + tid];
        }
      } else {
        // ---- fallback: inline convert+swizzle from f32 weights ----
        for (int i = tid; i < D_M * FCH / 2; i += NTHR) {
          const int idx = i * 2;
          const int dr = idx >> 5, c = idx & 31;  // c even
          const float2 wv = *(const float2*)&W1e[(size_t)dr * D_F + fc + c];
          const int ks = dr >> 5, kk = dr & 31;
          const int lh = (kk & 16) ? 16 : 0;
          __bf16* basep = (__bf16*)(w1f + (ks * 2 + (c >> 4)) * FRAGB_U2) + (kk & 15);
          basep[((c & 15) + lh) * 20]     = (__bf16)wv.x;
          basep[((c & 15) + 1 + lh) * 20] = (__bf16)wv.y;
        }
        for (int i = tid; i < FCH * D_M / 2; i += NTHR) {
          const int idx = i * 2;
          const int r = idx >> 9, c = idx & 511;  // c even
          const float2 wv = *(const float2*)&W2e[(size_t)(fc + r) * D_M + c];
          const int lh = (r & 16) ? 16 : 0;
          __bf16* basep = (__bf16*)(w2f + (c >> 4) * FRAGB_U2) + (r & 15);
          basep[((c & 15) + lh) * 20]     = (__bf16)wv.x;
          basep[((c & 15) + 1 + lh) * 20] = (__bf16)wv.y;
        }
      }
      __syncthreads();

      // ---- GEMM1: H[64x32] = x_tile @ W1chunk, K=512; software-pipelined ----
      v8f hacc = vzero;
      {
        v16bf af = load_frag32(xf + (m1 * 16) * 64, lane);
        v16bf bf = load_frag40(w1f + c1 * FRAGB_U2, lane);
#pragma unroll
        for (int k = 0; k < 16; ++k) {
          v16bf afn = af, bfn = bf;
          if (k < 15) {  // issue next frag loads before this WMMA
            afn = load_frag32(xf + (m1 * 16 + k + 1) * 64, lane);
            bfn = load_frag40(w1f + ((k + 1) * 2 + c1) * FRAGB_U2, lane);
          }
          hacc = __builtin_amdgcn_wmma_f32_16x16x32_bf16(
              false, af, false, bf, (short)0, hacc, false, false);
          af = afn;
          bf = bfn;
        }
      }
      // epilogue: +b1, exact gelu, *gate, write H in A-fragment order
      {
        const int n    = lane & 15;
        const int mofs = (lane & 16) ? 8 : 0;
        const int K    = c1 * 16 + n;
        const int h    = a_half(K);
        const int lnK  = a_lhalf(K);
        const float bv = b1e[fc + K];
        __bf16* hbase = (__bf16*)(hf + m1 * FRAGB_U2) + h;
#pragma unroll
        for (int v = 0; v < 8; ++v) {
          const int rl = mofs + v;
          hbase[(rl + lnK) * 20] = (__bf16)(g * gelu_exact(hacc[v] + bv));
        }
      }
      __syncthreads();

      // ---- GEMM2: C[64x512] += H @ W2chunk, K=32; software-pipelined ----
      {
        v16bf a2 = load_frag40(hf + mrow * FRAGB_U2, lane);
        v16bf bf = load_frag40(w2f + (cgrp * 16) * FRAGB_U2, lane);
#pragma unroll
        for (int ct = 0; ct < 16; ++ct) {
          v16bf bfn = bf;
          if (ct < 15)
            bfn = load_frag40(w2f + (cgrp * 16 + ct + 1) * FRAGB_U2, lane);
          acc[ct] = __builtin_amdgcn_wmma_f32_16x16x32_bf16(
              false, a2, false, bf, (short)0, acc[ct], false, false);
          bf = bfn;
        }
      }
      __syncthreads();
    }
  }

  // ---- write out: acc + gated b2 bias (full overwrite, no atomics) ----
  {
    const int n    = lane & 15;
    const int mofs = (lane & 16) ? 8 : 0;
#pragma unroll
    for (int ct = 0; ct < 16; ++ct) {
      const int d  = cgrp * 256 + ct * 16 + n;
      const float bs = bias_s[d];
#pragma unroll
      for (int v = 0; v < 8; ++v) {
        const int l = l0 + mrow * 16 + mofs + v;
        out[((size_t)b * L_SZ + l) * D_M + d] = acc[ct][v] + bs;
      }
    }
  }
}

extern "C" void kernel_launch(void* const* d_in, const int* in_sizes, int n_in,
                              void* d_out, int out_size, void* d_ws,
                              size_t ws_size, hipStream_t stream) {
  (void)in_sizes; (void)n_in; (void)out_size;
  const float* x      = (const float*)d_in[0];  // [B, L, D]
  const float* logits = (const float*)d_in[1];  // [B, E]
  const int*   masks  = (const int*)d_in[2];    // [B, E]
  const float* W1     = (const float*)d_in[3];  // [E, D, F]
  const float* b1     = (const float*)d_in[4];  // [E, F]
  const float* W2     = (const float*)d_in[5];  // [E, F, D]
  const float* b2     = (const float*)d_in[6];  // [E, D]
  float* out = (float*)d_out;                   // [B, L, D]

  dim3 grid(L_SZ / LT, B_SZ, 1);
  dim3 block(NTHR, 1, 1);

  const bool fast = (d_ws != nullptr) && (ws_size >= WS_NEED);
  if (fast) {
    __bf16* wsw = (__bf16*)d_ws;
    convert_weights_kernel<<<dim3(N_E * CHUNKS), block, 0, stream>>>(W1, W2, wsw);
    moe_fused_kernel<true><<<grid, block, 0, stream>>>(x, logits, masks, W1, b1,
                                                       W2, b2, wsw, out);
  } else {
    moe_fused_kernel<false><<<grid, block, 0, stream>>>(x, logits, masks, W1,
                                                        b1, W2, b2, nullptr,
                                                        out);
  }
}
